// Deepset_39968965657065
// MI455X (gfx1250) — matrix operand order repaired and checked by reading, not compiled, not run on hardware
//
#include <hip/hip_runtime.h>

// MI455X (gfx1250, wave32). Memory-bound GEMM pipeline:
//   HBM floor = 1.64GB / 23.3TB/s ~= 70us. bf16 hi/lo split (3x v_wmma_f32_16x16x32_bf16
//   per K=32) keeps near-fp32 accuracy while staying under the HBM roofline.

typedef __attribute__((ext_vector_type(16))) __bf16      v16bf;
typedef __attribute__((ext_vector_type(8)))  float       v8f;
typedef __attribute__((ext_vector_type(8)))  unsigned int v8u;

#define D_FEAT 1024
#define HIDDEN 64
#define NGRAPH 512

// Split two fp32 into packed bf16 hi parts (.x) and packed bf16 lo residual parts (.y).
// Low 16 bits of each dword = first (even-K) element, matching WMMA 16-bit layouts.
__device__ __forceinline__ uint2 split2(float x0, float x1) {
    unsigned b0 = __float_as_uint(x0), b1 = __float_as_uint(x1);
    unsigned h0 = b0 & 0xffff0000u,    h1 = b1 & 0xffff0000u;
    float r0 = x0 - __uint_as_float(h0);
    float r1 = x1 - __uint_as_float(h1);
    uint2 r;
    r.x = h1 | (h0 >> 16);
    r.y = (__float_as_uint(r1) & 0xffff0000u) | (__float_as_uint(r0) >> 16);
    return r;
}

__device__ __forceinline__ v8f wmma_bf16(v16bf a, v16bf b, v8f c) {
    return __builtin_amdgcn_wmma_f32_16x16x32_bf16(false, a, false, b, (short)0, c, false, false);
}

__device__ __forceinline__ float half_swap(float v) {   // SWAPX16 across wave32 halves
    return __int_as_float(__builtin_amdgcn_ds_swizzle(__float_as_int(v), 0x401F));
}

// Convert 16 fp32 (A-fragment order) into hi/lo bf16 fragments.
__device__ __forceinline__ void build_afrag(float4 a0, float4 a1, float4 a2, float4 a3,
                                            v16bf& Ah, v16bf& Al) {
    v8u hu, lu;
    uint2 r;
    r = split2(a0.x, a0.y); hu[0] = r.x; lu[0] = r.y;
    r = split2(a0.z, a0.w); hu[1] = r.x; lu[1] = r.y;
    r = split2(a1.x, a1.y); hu[2] = r.x; lu[2] = r.y;
    r = split2(a1.z, a1.w); hu[3] = r.x; lu[3] = r.y;
    r = split2(a2.x, a2.y); hu[4] = r.x; lu[4] = r.y;
    r = split2(a2.z, a2.w); hu[5] = r.x; lu[5] = r.y;
    r = split2(a3.x, a3.y); hu[6] = r.x; lu[6] = r.y;
    r = split2(a3.z, a3.w); hu[7] = r.x; lu[7] = r.y;
    Ah = __builtin_bit_cast(v16bf, hu);
    Al = __builtin_bit_cast(v16bf, lu);
}

// ---------------------------------------------------------------------------
// Kernel 1: zero accumulators, pre-split W1/W2 into pair-packed bf16 hi/lo.
// W*hp[p*64+n] packs bf16(W[2p][n]) in [15:0] and bf16(W[2p+1][n]) in [31:16].
// ---------------------------------------------------------------------------
__global__ __launch_bounds__(256) void gnn_prep(const float* __restrict__ W1,
                                                const float* __restrict__ W2,
                                                unsigned* __restrict__ W1hp, unsigned* __restrict__ W1lp,
                                                unsigned* __restrict__ W2hp, unsigned* __restrict__ W2lp,
                                                float* __restrict__ sums, float* __restrict__ cnt) {
    int tid = blockIdx.x * 256 + threadIdx.x;       // 0..32767
    sums[tid] = 0.0f;
    if (tid < NGRAPH) cnt[tid] = 0.0f;
    {
        int p = tid >> 6, n = tid & 63;             // p in [0,512)
        float x0 = W1[(2 * p) * HIDDEN + n];
        float x1 = W1[(2 * p + 1) * HIDDEN + n];
        uint2 r = split2(x0, x1);
        W1hp[tid] = r.x; W1lp[tid] = r.y;
    }
    if (tid < 2048) {                               // 32x64 for W2
        int p = tid >> 6, n = tid & 63;
        float x0 = W2[(2 * p) * HIDDEN + n];
        float x1 = W2[(2 * p + 1) * HIDDEN + n];
        uint2 r = split2(x0, x1);
        W2hp[tid] = r.x; W2lp[tid] = r.y;
    }
}

// ---------------------------------------------------------------------------
// Kernel 2: h2 = (relu(x@W1+b1))@W2 + b2, segment-sum into sums/cnt.
// 8 waves/block, each wave owns 16 rows x all 64 cols.
// ---------------------------------------------------------------------------
__global__ __launch_bounds__(256) void gnn_main(const float* __restrict__ x,
                                                const int*   __restrict__ batch,
                                                const float* __restrict__ b1,
                                                const float* __restrict__ b2,
                                                const unsigned* __restrict__ W1hp,
                                                const unsigned* __restrict__ W1lp,
                                                const unsigned* __restrict__ W2hp,
                                                const unsigned* __restrict__ W2lp,
                                                float* __restrict__ sums,
                                                float* __restrict__ cnt) {
    __shared__ float h1s[8][16][68];                // per-wave 16x64 tile, padded

    const int wave = threadIdx.x >> 5;
    const int lane = threadIdx.x & 31;
    const int m    = lane & 15;                     // M (A/C row) or N (B/C col) index
    const int hf   = lane >> 4;                     // lane half selects K range
    const int rb   = (blockIdx.x * 8 + wave) * 16;  // this wave's row base

    const float* xrow = x + (size_t)(rb + m) * D_FEAT + hf * 16;

    // ---------------- layer 1: C = x_tile @ W1 ----------------
    v8f C[4] = {};
    for (int k0 = 0; k0 < D_FEAT; k0 += 32) {
        float4 a0 = *(const float4*)(xrow + k0);
        float4 a1 = *(const float4*)(xrow + k0 + 4);
        float4 a2 = *(const float4*)(xrow + k0 + 8);
        float4 a3 = *(const float4*)(xrow + k0 + 12);
        __builtin_prefetch(xrow + k0 + 128, 0, 1);  // stream-ahead (global_prefetch)
        v16bf Ah, Al;
        build_afrag(a0, a1, a2, a3, Ah, Al);

        const int bidx0 = (k0 >> 1) * 64 + hf * 512 + m;   // p = k0/2 + 8*hf + j
#pragma unroll
        for (int c = 0; c < 4; ++c) {
            v8u bh, bl;
            const int bi = bidx0 + c * 16;
#pragma unroll
            for (int j = 0; j < 8; ++j) {
                bh[j] = W1hp[bi + j * 64];
                bl[j] = W1lp[bi + j * 64];
            }
            v16bf Bh = __builtin_bit_cast(v16bf, bh);
            v16bf Bl = __builtin_bit_cast(v16bf, bl);
            C[c] = wmma_bf16(Ah, Bh, C[c]);
            C[c] = wmma_bf16(Ah, Bl, C[c]);
            C[c] = wmma_bf16(Al, Bh, C[c]);
        }
    }

    // bias + relu, stash tile in LDS to re-layout C(row-in-vgpr) -> A(row-in-lane)
#pragma unroll
    for (int c = 0; c < 4; ++c) {
        float bb = b1[c * 16 + m];
#pragma unroll
        for (int g = 0; g < 8; ++g) {
            h1s[wave][g + 8 * hf][c * 16 + m] = fmaxf(C[c][g] + bb, 0.0f);
        }
    }
    // intra-wave LDS RAW only: DScnt ordering, no workgroup barrier needed.

    // ---------------- layer 2: D = h1 @ W2 ----------------
    v8f D[4] = {};
#pragma unroll
    for (int kc = 0; kc < 2; ++kc) {
        const float* hrow = &h1s[wave][m][kc * 32 + 16 * hf];
        float4 a0 = *(const float4*)(hrow);
        float4 a1 = *(const float4*)(hrow + 4);
        float4 a2 = *(const float4*)(hrow + 8);
        float4 a3 = *(const float4*)(hrow + 12);
        v16bf Ah, Al;
        build_afrag(a0, a1, a2, a3, Ah, Al);

        const int bidx0 = kc * 1024 + hf * 512 + m;        // p = kc*16 + 8*hf + j
#pragma unroll
        for (int c = 0; c < 4; ++c) {
            v8u bh, bl;
            const int bi = bidx0 + c * 16;
#pragma unroll
            for (int j = 0; j < 8; ++j) {
                bh[j] = W2hp[bi + j * 64];
                bl[j] = W2lp[bi + j * 64];
            }
            v16bf Bh = __builtin_bit_cast(v16bf, bh);
            v16bf Bl = __builtin_bit_cast(v16bf, bl);
            D[c] = wmma_bf16(Ah, Bh, D[c]);
            D[c] = wmma_bf16(Ah, Bl, D[c]);
            D[c] = wmma_bf16(Al, Bh, D[c]);
        }
    }
#pragma unroll
    for (int c = 0; c < 4; ++c) {
        float bb = b2[c * 16 + m];
#pragma unroll
        for (int g = 0; g < 8; ++g) D[c][g] += bb;
    }

    // ---------------- segment accumulation (batch is sorted) ----------------
    int gFirst = batch[rb];
    int gLast  = batch[rb + 15];
    if (gFirst == gLast) {
        // whole 16-row tile belongs to one graph: reduce rows, 1 atomic per column
#pragma unroll
        for (int c = 0; c < 4; ++c) {
            float part = 0.0f;
#pragma unroll
            for (int g = 0; g < 8; ++g) part += D[c][g];
            float s = part + half_swap(part);              // add other lane-half's 8 rows
            if (lane < 16) atomicAdd(&sums[gFirst * HIDDEN + c * 16 + m], s);
        }
        if (lane == 0) atomicAdd(&cnt[gFirst], 16.0f);
    } else {
        // graph boundary inside tile (rare): per-row atomics
#pragma unroll
        for (int g = 0; g < 8; ++g) {
            int bg = batch[rb + g + 8 * hf];
#pragma unroll
            for (int c = 0; c < 4; ++c)
                atomicAdd(&sums[bg * HIDDEN + c * 16 + m], D[c][g]);
            if (m == 0) atomicAdd(&cnt[bg], 1.0f);
        }
    }
}

// ---------------------------------------------------------------------------
// Kernel 3: pooled = sums/max(cnt,1); z = (pooled@W3+b3)@W4+b4; softmax(dim=0).
// One block, 512 threads (1 thread == 1 graph).
// ---------------------------------------------------------------------------
__global__ __launch_bounds__(512) void gnn_head(const float* __restrict__ sums,
                                                const float* __restrict__ cnt,
                                                const float* __restrict__ W3,
                                                const float* __restrict__ b3,
                                                const float* __restrict__ W4,
                                                const float* __restrict__ b4,
                                                float* __restrict__ out) {
    __shared__ float w3s[64 * 32];
    __shared__ float w4s[64];
    __shared__ float b3s[32];
    __shared__ float b4s[2];
    __shared__ float zb[2][NGRAPH];
    __shared__ float eb[2][NGRAPH];
    __shared__ float red[4];

    const int t = threadIdx.x;
    for (int i = t; i < 2048; i += 512) w3s[i] = W3[i];
    if (t < 64) w4s[t] = W4[t];
    if (t < 32) b3s[t] = b3[t];
    if (t < 2)  b4s[t] = b4[t];
    __syncthreads();

    float inv = 1.0f / fmaxf(cnt[t], 1.0f);
    float z1[32];
#pragma unroll
    for (int j = 0; j < 32; ++j) z1[j] = b3s[j];
    for (int i = 0; i < 64; ++i) {
        float p = sums[t * 64 + i] * inv;
#pragma unroll
        for (int j = 0; j < 32; ++j) z1[j] += p * w3s[i * 32 + j];
    }
    float z2[2];
#pragma unroll
    for (int c = 0; c < 2; ++c) {
        float a = b4s[c];
#pragma unroll
        for (int j = 0; j < 32; ++j) a += z1[j] * w4s[j * 2 + c];
        z2[c] = a;
        zb[c][t] = a;
    }
    __syncthreads();
    if (t < 2) {
        float mx = zb[t][0];
        for (int g = 1; g < NGRAPH; ++g) mx = fmaxf(mx, zb[t][g]);
        red[t] = mx;
    }
    __syncthreads();
    float e[2];
#pragma unroll
    for (int c = 0; c < 2; ++c) {
        e[c] = __expf(z2[c] - red[c]);
        eb[c][t] = e[c];
    }
    __syncthreads();
    if (t < 2) {
        float s = 0.0f;
        for (int g = 0; g < NGRAPH; ++g) s += eb[t][g];
        red[2 + t] = s;
    }
    __syncthreads();
#pragma unroll
    for (int c = 0; c < 2; ++c) out[t * 2 + c] = e[c] / red[2 + c];
}

// ---------------------------------------------------------------------------
extern "C" void kernel_launch(void* const* d_in, const int* in_sizes, int n_in,
                              void* d_out, int out_size, void* d_ws, size_t ws_size,
                              hipStream_t stream) {
    (void)in_sizes; (void)n_in; (void)out_size; (void)ws_size;
    const float* x     = (const float*)d_in[0];
    const int*   batch = (const int*)d_in[1];
    const float* W1    = (const float*)d_in[2];
    const float* b1    = (const float*)d_in[3];
    const float* W2    = (const float*)d_in[4];
    const float* b2    = (const float*)d_in[5];
    const float* W3    = (const float*)d_in[6];
    const float* b3    = (const float*)d_in[7];
    const float* W4    = (const float*)d_in[8];
    const float* b4    = (const float*)d_in[9];
    float* out = (float*)d_out;

    // workspace layout (~402 KB total)
    char* ws = (char*)d_ws;
    float*    sums = (float*)(ws);                 // 512*64 f32   (128 KB)
    float*    cnt  = (float*)(ws + 131072);        // 512 f32
    unsigned* W1hp = (unsigned*)(ws + 133120);     // 512*64 u32   (128 KB)
    unsigned* W1lp = (unsigned*)(ws + 264192);     // 512*64 u32
    unsigned* W2hp = (unsigned*)(ws + 395264);     // 32*64 u32
    unsigned* W2lp = (unsigned*)(ws + 403456);     // 32*64 u32

    gnn_prep<<<128, 256, 0, stream>>>(W1, W2, W1hp, W1lp, W2hp, W2lp, sums, cnt);
    gnn_main<<<3125, 256, 0, stream>>>(x, batch, b1, b2, W1hp, W1lp, W2hp, W2lp, sums, cnt);
    gnn_head<<<1, 512, 0, stream>>>(sums, cnt, W3, b3, W4, b4, out);
}
